// BasalGanglia_62225486184904
// MI455X (gfx1250) — matrix operand order repaired and checked by reading, not compiled, not run on hardware
//
#include <hip/hip_runtime.h>
#include <hip/hip_bf16.h>
#include <math.h>

// ---------------- problem constants ----------------
#define BATCH   128
#define CTX     4096
#define DD      2048
#define NGPI    2
#define FFSTEPS 20
#define NITER   50
#define ETA_GPE 1.0f
#define ETA_STN (1.0f/3.0f)
#define ETA_GPI 0.1f
#define WSG     2.0f
#define WGS     (-2.0f)
#define EPS_L   0.005f

typedef __attribute__((ext_vector_type(16))) __bf16        v16bf;
typedef __attribute__((ext_vector_type(8)))  float         v8f;
typedef __attribute__((ext_vector_type(4)))  unsigned int  v4ui;
typedef __attribute__((ext_vector_type(8)))  int           v8i;
typedef __attribute__((ext_vector_type(4)))  int           v4i;

#if defined(__has_builtin)
#if __has_builtin(__builtin_amdgcn_tensor_load_to_lds) && __has_builtin(__builtin_amdgcn_s_wait_tensorcnt)
#define HAVE_TDM 1
#endif
#endif

// =====================================================================
// TDM issue: DMA a (rows=4) x (cols=128) f32 slice of W (row pitch DD)
// into LDS at lds_byte, packed row-major (pitch 128 floats, no pad).
// Descriptor per CDNA5 ISA 8.3/8.4 (D# group0 128b, group1 256b).
// =====================================================================
#ifdef HAVE_TDM
__device__ __forceinline__ void tdm_load_slice(const float* gsrc, unsigned lds_byte)
{
    unsigned long long ga = (unsigned long long)(const void*)gsrc;
    v4ui g0;
    g0.x = 1u;                                  // count=1, user-mode, no gather
    g0.y = lds_byte;                            // lds_addr (bytes, wave-relative)
    g0.z = (unsigned)ga;                        // global_addr[31:0]
    g0.w = (unsigned)(ga >> 32) | (2u << 30);   // global_addr[56:32] | type=2 ("image")
    v8i g1;
    g1[0] = (int)(2u << 16);                    // wg_mask=0 | data_size=2 (4B) | no pad/iter
    g1[1] = (int)((DD & 0xFFFF) << 16);         // abar=0 | tensor_dim0[15:0]=2048
    g1[2] = (int)((DD >> 16) | ((CTX & 0xFFFF) << 16));   // dim0 hi | tensor_dim1 lo (4096)
    g1[3] = (int)((CTX >> 16) | (128u << 16));  // dim1 hi | tile_dim0=128
    g1[4] = 4;                                  // tile_dim1=4 | tile_dim2=0
    g1[5] = DD;                                 // tensor_dim0_stride[31:0]=2048
    g1[6] = 0;                                  // stride hi | tensor_dim1_stride lo (unused, 2D)
    g1[7] = 0;
    v4i z4 = {0, 0, 0, 0};                      // groups 2/3 unused (2D tensor)
#if __clang_major__ >= 23
    v8i z8 = {0, 0, 0, 0, 0, 0, 0, 0};
    __builtin_amdgcn_tensor_load_to_lds(g0, g1, z4, z4, z8, 0);
#else
    __builtin_amdgcn_tensor_load_to_lds(g0, g1, z4, z4, 0);
#endif
}
#endif

// =====================================================================
// Kernel 1: four fused GEMMs  J/K = stimulus @ W   (bf16 WMMA, f32 acc)
// block = 256 thr (8 wave32); tile = 16(M) x 128(N); K-chunk = 32
// B staged via TDM (double-buffered, overlapped with WMMA); A via b128.
// =====================================================================
#define AP  36              // padded LDS pitch for A (floats)
#define NCH (CTX / 32)      // 128 K-chunks

__global__ __launch_bounds__(256) void bg_gemm_kernel(
    const float* __restrict__ stim,
    const float* __restrict__ w0, const float* __restrict__ w1,
    const float* __restrict__ w2, const float* __restrict__ w3,
    float* __restrict__ outbase)
{
    __shared__ float As[2][16 * AP];     // A tile 16x32, row-major, padded
    __shared__ float Bs[2][32 * 128];    // B tile 32x128, packed [k][n] (TDM layout)

    const int t    = threadIdx.x;
    const int lane = t & 31;
    const int wv   = t >> 5;
    const int n0   = blockIdx.x * 128;
    const int m0   = blockIdx.y * 16;
    const int z    = blockIdx.z;

    const float* W = (z == 0) ? w0 : (z == 1) ? w1 : (z == 2) ? w2 : w3;
    float* out = outbase + (size_t)z * BATCH * DD;

    v8f acc = {};
    const int mA    = lane & 15;              // A row
    const int khalf = lane >> 4;              // K-half select
    const int nl    = wv * 16 + (lane & 15);  // local N column (0..127)

    // ---- staging helpers ----
    auto stageA = [&](int k0, int buf) {
        if (t < 128) {
            int m = t >> 3, kq = t & 7;
            const float4 v = *(const float4*)(stim + (size_t)(m0 + m) * CTX + k0 + kq * 4);
            *(float4*)(&As[buf][m * AP + kq * 4]) = v;   // 144B pitch: 16B aligned
        }
    };
    auto stageB = [&](int k0, int buf) {
#ifdef HAVE_TDM
        // each wave DMAs its own 4-row slice; per-wave-uniform descriptor -> SGPRs
        const float* src = W + (size_t)(k0 + 4 * wv) * DD + n0;
        unsigned lds = (unsigned)(unsigned long long)(&Bs[buf][wv * 4 * 128]);
        tdm_load_slice(src, lds);
#else
        #pragma unroll
        for (int i = 0; i < 4; ++i) {
            int idx = i * 256 + t;
            int k   = idx >> 5;
            int nq  = idx & 31;
            *(float4*)(&Bs[buf][k * 128 + nq * 4]) =
                *(const float4*)(W + (size_t)(k0 + k) * DD + n0 + nq * 4);
        }
#endif
    };

    // ---- prologue: fill buffer 0 ----
    stageA(0, 0);
    stageB(0, 0);
#ifdef HAVE_TDM
    __builtin_amdgcn_s_wait_tensorcnt(0);
#endif
    __syncthreads();

    // ---- main pipeline: DMA chunk i+1 while computing chunk i ----
    for (int i = 0; i < NCH; ++i) {
        const int cur = i & 1, nxt = cur ^ 1;
        if (i + 1 < NCH) {
            stageA((i + 1) * 32, nxt);
            stageB((i + 1) * 32, nxt);
        }

        // fragments per ISA VGPR layout, f32 -> bf16 in-register
        v16bf afrag, bfrag;
        const float* Ar = &As[cur][mA * AP + khalf * 8];
        #pragma unroll
        for (int e = 0; e < 8; ++e) afrag[e]     = (__bf16)Ar[e];       // K = 8*kh + e
        #pragma unroll
        for (int e = 0; e < 8; ++e) afrag[8 + e] = (__bf16)Ar[16 + e];  // K = 16 + 8*kh + e
        const float* Bc = &Bs[cur][khalf * 8 * 128 + nl];
        #pragma unroll
        for (int e = 0; e < 8; ++e) bfrag[e] = (__bf16)Bc[e * 128];    // K = 8*kh + e
        #pragma unroll
        for (int e = 0; e < 8; ++e) bfrag[8 + e] = (__bf16)Bc[(16 + e) * 128];  // K = 16 + 8*kh + e

        acc = __builtin_amdgcn_wmma_f32_16x16x32_bf16(
                  false, afrag, false, bfrag, (short)0, acc, false, false);

#ifdef HAVE_TDM
        __builtin_amdgcn_s_wait_tensorcnt(0);    // my next-chunk DMA done (overlapped)
#endif
        __syncthreads();                          // everyone done reading cur / writing nxt
    }

    // ---- epilogue: C layout (VGPR v: lanes 0-15 -> M=v, lanes 16-31 -> M=v+8) ----
    #pragma unroll
    for (int v = 0; v < 8; ++v) {
        int m = (lane < 16) ? v : v + 8;
        out[(size_t)(m0 + m) * DD + n0 + nl] = acc[v];
    }
}

// =====================================================================
// Kernel 2: 20-step FF recurrence (V = J + V*(1-J-K)), V_GPi_DP, lamd2
// one block per batch row; 256 thr x 8 elems
// =====================================================================
__global__ __launch_bounds__(256) void bg_ff_dp_kernel(
    const float* __restrict__ jk,       // J_D1,K_D1,J_D2,K_D2 each BATCH*DD
    const float* __restrict__ wd1gpi,   // (DD,2)
    const float* __restrict__ deltavf,  // (BATCH)
    float* __restrict__ vd2_out,        // (BATCH,DD)
    float* __restrict__ dp_out,         // (BATCH,2)
    float* __restrict__ lam_out)        // (BATCH)
{
    const int b = blockIdx.x, t = threadIdx.x;
    const float* J1 = jk;
    const float* K1 = jk + 1 * (size_t)BATCH * DD;
    const float* J2 = jk + 2 * (size_t)BATCH * DD;
    const float* K2 = jk + 3 * (size_t)BATCH * DD;

    float s0 = 0.f, s1 = 0.f;
    #pragma unroll
    for (int i = 0; i < 8; ++i) {
        int j = t + i * 256;
        float j1 = J1[(size_t)b * DD + j], k1 = K1[(size_t)b * DD + j];
        float j2 = J2[(size_t)b * DD + j], k2 = K2[(size_t)b * DD + j];
        float r1 = 1.f - j1 - k1, r2 = 1.f - j2 - k2;
        float v1 = 0.f, v2 = 0.f;
        #pragma unroll
        for (int s = 0; s < FFSTEPS; ++s) { v1 = j1 + v1 * r1; v2 = j2 + v2 * r2; }
        vd2_out[(size_t)b * DD + j] = v2;
        s0 += v1 * wd1gpi[j * 2 + 0];
        s1 += v1 * wd1gpi[j * 2 + 1];
    }
    __shared__ float2 red[256];
    red[t] = {s0, s1};
    __syncthreads();
    for (int off = 128; off; off >>= 1) {
        if (t < off) { red[t].x += red[t + off].x; red[t].y += red[t + off].y; }
        __syncthreads();
    }
    if (t == 0) {
        dp_out[b * 2 + 0] = red[0].x;
        dp_out[b * 2 + 1] = red[0].y;
        lam_out[b] = 1.f / (1.f + __expf(-deltavf[b]));   // sigmoid(A2*(dv-theta)), A2=1, theta=0
    }
}

// =====================================================================
// Kernel 3: 50-iter STN/GPe/GPi dynamics.
// wlat_gpe = eps*(1-I)      -> x@wlat = eps*(rowsum(x) - x)
// wlat_stn = (1+eps)*1-eps*I -> v@wlat = (1+eps)*rowsum(v) - eps*v
// one block per batch row; state in registers; LDS tree reductions
// =====================================================================
__global__ __launch_bounds__(256) void bg_stn_kernel(
    const float* __restrict__ vd2,
    const float* __restrict__ wd2gpi,
    const float* __restrict__ dp,
    const float* __restrict__ lam,
    float* __restrict__ out)
{
    const int b = blockIdx.x, t = threadIdx.x;
    float v2l[8], w0l[8], w1l[8];
    #pragma unroll
    for (int i = 0; i < 8; ++i) {
        int j = t + i * 256;
        v2l[i] = vd2[(size_t)b * DD + j];
        w0l[i] = wd2gpi[j * 2 + 0];
        w1l[i] = wd2gpi[j * 2 + 1];
    }
    const float lm  = lam[b];
    const float dp0 = dp[b * 2 + 0], dp1 = dp[b * 2 + 1];

    float xg[8] = {}, xs[8] = {}, vs[8] = {};
    float g0 = 0.f, g1 = 0.f;
    __shared__ float2 red[256];

    for (int it = 0; it < NITER; ++it) {
        float sg = 0.f, sv = 0.f;
        #pragma unroll
        for (int i = 0; i < 8; ++i) { sg += xg[i]; sv += vs[i]; }
        red[t] = {sg, sv};
        __syncthreads();
        for (int off = 128; off; off >>= 1) {
            if (t < off) { red[t].x += red[t + off].x; red[t].y += red[t + off].y; }
            __syncthreads();
        }
        float SG = red[0].x, SV = red[0].y;
        __syncthreads();

        float p0 = 0.f, p1 = 0.f;
        #pragma unroll
        for (int i = 0; i < 8; ++i) {
            float xgo = xg[i];
            float xgn = xgo + ETA_GPE * (-xgo + WSG * vs[i] + EPS_L * (SG - xgo) - v2l[i]);
            float xsn = xs[i] + ETA_STN * (-xs[i] + WGS * xgn + (1.f + EPS_L) * SV - EPS_L * vs[i]);
            xg[i] = xgn; xs[i] = xsn;
            float v = tanhf(lm * xsn);
            vs[i] = v;
            p0 += v * w0l[i];
            p1 += v * w1l[i];
        }
        red[t] = {p0, p1};
        __syncthreads();
        for (int off = 128; off; off >>= 1) {
            if (t < off) { red[t].x += red[t + off].x; red[t].y += red[t + off].y; }
            __syncthreads();
        }
        float IP0 = lm * red[0].x, IP1 = lm * red[0].y;
        __syncthreads();

        g0 = g0 + ETA_GPI * (-g0 - dp0 + 2.f * IP0);
        g1 = g1 + ETA_GPI * (-g1 - dp1 + 2.f * IP1);
    }
    if (t == 0) { out[b * 2 + 0] = -g0; out[b * 2 + 1] = -g1; }
}

// =====================================================================
extern "C" void kernel_launch(void* const* d_in, const int* in_sizes, int n_in,
                              void* d_out, int out_size, void* d_ws, size_t ws_size,
                              hipStream_t stream) {
    const float* stim = (const float*)d_in[0];   // (128,4096)
    const float* dvf  = (const float*)d_in[1];   // (128,1)
    const float* wj1  = (const float*)d_in[2];   // (4096,2048)
    const float* wk1  = (const float*)d_in[3];
    const float* wj2  = (const float*)d_in[4];
    const float* wk2  = (const float*)d_in[5];
    const float* wd1  = (const float*)d_in[6];   // (2048,2)
    const float* wd2  = (const float*)d_in[7];   // (2048,2)

    float* ws   = (float*)d_ws;
    float* jk   = ws;                                   // 4 * 128*2048
    float* vd2  = ws + 4 * (size_t)BATCH * DD;          // 128*2048
    float* lamb = ws + 5 * (size_t)BATCH * DD;          // 128
    float* dpb  = ws + 5 * (size_t)BATCH * DD + BATCH;  // 256

    dim3 g1(DD / 128, BATCH / 16, 4);
    bg_gemm_kernel<<<g1, 256, 0, stream>>>(stim, wj1, wk1, wj2, wk2, jk);
    bg_ff_dp_kernel<<<BATCH, 256, 0, stream>>>(jk, wd1, dvf, vd2, dpb, lamb);
    bg_stn_kernel<<<BATCH, 256, 0, stream>>>(vd2, wd2, dpb, lamb, (float*)d_out);
}
